// GTN_69148973466362
// MI455X (gfx1250) — matrix operand order repaired and checked by reading, not compiled, not run on hardware
//
#include <hip/hip_runtime.h>

// ---------------------------------------------------------------------------
// GTN forward for MI455X (gfx1250, wave32, WMMA + TDM).
// All matmuls (2 x 4096^3 GT products + GCN layers) go through one templated
// bf16 WMMA GEMM (v_wmma_f32_16x16x32_bf16, f32 accumulate):
//   - macro tile 128x128, BK=64, 8 waves (4x2), wave tile 32x64
//   - 16 WMMAs per barrier pair, ds_load_b128 fragment loads
//   - Tensor Data Mover staging: one tensor_load_to_lds per 128x64 tile,
//     double-buffered (wave 0 issues next tile while all waves compute),
//     hardware LDS row padding 128B+16B == our 72-element pitch
//   - compile-time epilogue (bias + branchless lrelu/relu, selectable
//     f32 / bf16 / bf16-transposed outputs)
// O(N^2) glue kernels handle gtconv softmax-sums, degree norms, adjacency,
// conversions, and the classification head.
// ---------------------------------------------------------------------------

typedef __attribute__((ext_vector_type(16))) __bf16 v16bf;
typedef __attribute__((ext_vector_type(8)))  __bf16 v8bf;
typedef __attribute__((ext_vector_type(8)))  float  v8f;
typedef __attribute__((ext_vector_type(4)))  unsigned int v4u;
typedef __attribute__((ext_vector_type(8)))  int  v8i;
typedef __attribute__((ext_vector_type(4)))  int  v4i;

#define GN 4096
#define GE 5

#if defined(__has_builtin)
#if __has_builtin(__builtin_amdgcn_tensor_load_to_lds) && \
    __has_builtin(__builtin_amdgcn_s_wait_tensorcnt)
#define USE_TDM 1
#endif
#endif
#ifndef USE_TDM
#define USE_TDM 0
#endif

// ------------------------------- GEMM --------------------------------------
#define BM 128
#define BN 128
#define BK 64
#define LP 72   // padded LDS pitch in bf16 elements (144B rows, 16B aligned)

#if USE_TDM
// Issue one TDM 2-D tile load: rows x BK bf16 elements, row stride K elements,
// into LDS at lds_byte_addr with 16B padding after each 128B row (pitch 144B).
// D# fields per CDNA5 ISA 8.3/8.4 (group0: count/lds/global/type,
// group1: data_size=2B, pad_interval=32 DW, pad_amount=4 DW, dims/strides).
// This toolchain's builtin takes 6 args (g0, g1, g2, g3, g-extra, cpol).
__device__ __forceinline__ void tdm_load_tile(const __bf16* gsrc,
                                              unsigned int lds_byte_addr,
                                              int rows, int K)
{
    unsigned long long ga = (unsigned long long)gsrc;
    v4u g0;
    g0[0] = 1u;                                            // count=1, user mode
    g0[1] = lds_byte_addr;                                 // lds_addr
    g0[2] = (unsigned int)ga;                              // global_addr[31:0]
    g0[3] = (unsigned int)((ga >> 32) & 0x1FFFFFFu)        // global_addr[56:32]
          | (2u << 30);                                    // type=2 (image)
    v8i g1;
    g1[0] = (1 << 16)      // data_size = 1 -> 2 bytes
          | (1 << 20)      // pad_enable
          | (4 << 22)      // pad_interval code 4 -> 32 DWORDs (128B rows)
          | (3 << 25);     // pad_amount  code 3 -> 4 DWORDs  (16B pad)
    g1[1] = (int)((unsigned)(K & 0xFFFF) << 16);           // tensor_dim0 lo16
    g1[2] = (int)(((unsigned)K >> 16)                      // tensor_dim0 hi16
          | ((unsigned)(rows & 0xFFFF) << 16));            // tensor_dim1 lo16
    g1[3] = (int)(((unsigned)rows >> 16)                   // tensor_dim1 hi16
          | ((unsigned)BK << 16));                         // tile_dim0 = 64
    g1[4] = rows & 0xFFFF;                                 // tile_dim1 (tile_dim2=0)
    g1[5] = K;                                             // tensor_dim0_stride lo32
    g1[6] = 0;                                             // stride0 hi16 / stride1 lo16
    g1[7] = 0;                                             // stride1 hi32
    v4i z4 = {0, 0, 0, 0};                                 // groups 2/3: 2-D tile
    v8i z8 = {0, 0, 0, 0, 0, 0, 0, 0};
    __builtin_amdgcn_tensor_load_to_lds(g0, g1, z4, z4, z8, 0);
}
#endif

template <int EPI, bool WF, bool WB, bool WBT>
__global__ __launch_bounds__(256)
void gemm_bf16_wmma(const __bf16* __restrict__ A, const __bf16* __restrict__ BT,
                    float* __restrict__ Cf, __bf16* __restrict__ Cb,
                    __bf16* __restrict__ CbT, const float* __restrict__ bias,
                    int M, int Nn, int K)
{
    __shared__ __align__(16) __bf16 As[2][BM * LP];
    __shared__ __align__(16) __bf16 Bs[2][BN * LP];

    const int tid  = threadIdx.x;
    const int lane = tid & 31;
    const int wid  = tid >> 5;          // 0..7
    const int wm   = wid & 3;           // wave row block (32 rows)
    const int wn   = wid >> 2;          // wave col block (64 cols)
    const int l16  = lane & 15;
    const int hi   = lane >> 4;         // lane half
    const int tileM = blockIdx.y * BM;
    const int tileN = blockIdx.x * BN;

    v8f acc[2][4] = {};

    const int nk = K / BK;

#if USE_TDM
    const __bf16* Abase = A  + (size_t)tileM * K;
    const __bf16* Bbase = BT + (size_t)tileN * K;
    // LDS aperture keeps the workgroup-relative LDS offset in addr[31:0].
    unsigned int asAddr[2] = { (unsigned int)(unsigned long long)(void*)&As[0][0],
                               (unsigned int)(unsigned long long)(void*)&As[1][0] };
    unsigned int bsAddr[2] = { (unsigned int)(unsigned long long)(void*)&Bs[0][0],
                               (unsigned int)(unsigned long long)(void*)&Bs[1][0] };
    if (wid == 0) {
        tdm_load_tile(Abase, asAddr[0], BM, K);
        tdm_load_tile(Bbase, bsAddr[0], BN, K);
    }
#endif

    for (int i = 0; i < nk; ++i) {
        const int cur = i & 1;
#if USE_TDM
        if (wid == 0) __builtin_amdgcn_s_wait_tensorcnt(0);  // tiles landed
        __syncthreads();                                     // publish to block
        if (wid == 0 && (i + 1) < nk) {                      // prefetch next
            tdm_load_tile(Abase + (size_t)(i + 1) * BK, asAddr[1 - cur], BM, K);
            tdm_load_tile(Bbase + (size_t)(i + 1) * BK, bsAddr[1 - cur], BN, K);
        }
#else
        const int ko = i * BK;
        #pragma unroll
        for (int it = 0; it < 4; ++it) {
            int idx = tid + it * 256;
            int r   = idx >> 3;
            int c8  = (idx & 7) * 8;
            *(v8bf*)(&As[cur][r * LP + c8]) =
                *(const v8bf*)(A + (size_t)(tileM + r) * K + ko + c8);
            *(v8bf*)(&Bs[cur][r * LP + c8]) =
                *(const v8bf*)(BT + (size_t)(tileN + r) * K + ko + c8);
        }
        __syncthreads();
#endif

        // ---- two K=32 slices per staged tile: 16 WMMAs per barrier pair ----
        #pragma unroll
        for (int ks = 0; ks < 2; ++ks) {
            // A frag: lane l -> row m=l%16; half selects K {0..7,16..23}/{8..15,24..31}
            v16bf aF[2];
            #pragma unroll
            for (int i2 = 0; i2 < 2; ++i2) {
                int m  = wm * 32 + i2 * 16 + l16;
                int kh = ks * 32 + (hi ? 8 : 0);
                union { v16bf v; v8bf h[2]; } u;
                u.h[0] = *(const v8bf*)(&As[cur][m * LP + kh]);
                u.h[1] = *(const v8bf*)(&As[cur][m * LP + kh + 16]);
                aF[i2] = u.v;
            }
            // B frag: lane l -> col n=l%16; half selects K {0..7,16..23}/{8..15,24..31}
            v16bf bF[4];
            #pragma unroll
            for (int j = 0; j < 4; ++j) {
                int n  = wn * 64 + j * 16 + l16;
                int kh = ks * 32 + (hi ? 8 : 0);
                union { v16bf v; v8bf h[2]; } u;
                u.h[0] = *(const v8bf*)(&Bs[cur][n * LP + kh]);
                u.h[1] = *(const v8bf*)(&Bs[cur][n * LP + kh + 16]);
                bF[j] = u.v;
            }
            #pragma unroll
            for (int i2 = 0; i2 < 2; ++i2)
                #pragma unroll
                for (int j = 0; j < 4; ++j)
                    acc[i2][j] = __builtin_amdgcn_wmma_f32_16x16x32_bf16(
                        false, aF[i2], false, bF[j], (short)0, acc[i2][j], false, false);
        }
        __syncthreads();   // all reads done before buffer is refilled at i+2
    }

    // ---- epilogue: bias hoisted per fragment column, branchless act ----
    float bn[4];
    #pragma unroll
    for (int j = 0; j < 4; ++j)
        bn[j] = (EPI != 0) ? bias[tileN + wn * 64 + j * 16 + l16] : 0.f;

    #pragma unroll
    for (int i = 0; i < 2; ++i)
        #pragma unroll
        for (int j = 0; j < 4; ++j)
            #pragma unroll
            for (int r = 0; r < 8; ++r) {
                int m = tileM + wm * 32 + i * 16 + (hi ? r + 8 : r);
                int n = tileN + wn * 64 + j * 16 + l16;
                float v = acc[i][j][r];
                if (EPI != 0) v += bn[j];
                if (EPI == 1) v = fmaxf(v, 0.f) + 0.25f * fminf(v, 0.f); // leaky
                if (EPI == 2) v = fmaxf(v, 0.f);                         // relu
                if (WF)  Cf [(size_t)m * Nn + n] = v;
                if (WB)  Cb [(size_t)m * Nn + n] = (__bf16)v;
                if (WBT) CbT[(size_t)n * M  + m] = (__bf16)v;
            }
}

// ------------------------- GTConv (fused x3) -------------------------------
__device__ __forceinline__ void softmax5(const float* __restrict__ w, float* s) {
    float mx = w[0];
    #pragma unroll
    for (int e = 1; e < GE; ++e) mx = fmaxf(mx, w[e]);
    float sm = 0.f;
    #pragma unroll
    for (int e = 0; e < GE; ++e) { s[e] = __expf(w[e] - mx); sm += s[e]; }
    float r = 1.f / sm;
    #pragma unroll
    for (int e = 0; e < GE; ++e) s[e] *= r;
}

__global__ void gtconv3_kernel(const float* __restrict__ A,
                               const float* __restrict__ w0,
                               const float* __restrict__ w1,
                               const float* __restrict__ w2,
                               __bf16* __restrict__ aOut,    // a   row-major
                               __bf16* __restrict__ bTOut,   // b^T row-major
                               __bf16* __restrict__ a1TOut)  // a1^T row-major
{
    size_t idx = (size_t)blockIdx.x * blockDim.x + threadIdx.x;
    if (idx >= (size_t)GN * GN) return;
    int n = (int)(idx / GN), m = (int)(idx % GN);
    float s0[GE], s1[GE], s2[GE];
    softmax5(w0, s0); softmax5(w1, s1); softmax5(w2, s2);
    const float* p = A + idx * GE;
    float v0 = 0.f, v1 = 0.f, v2 = 0.f;
    #pragma unroll
    for (int e = 0; e < GE; ++e) {
        float ae = p[e];
        v0 += s0[e] * ae; v1 += s1[e] * ae; v2 += s2[e] * ae;
    }
    aOut  [(size_t)n * GN + m] = (__bf16)v0;
    bTOut [(size_t)m * GN + n] = (__bf16)v1;
    a1TOut[(size_t)m * GN + n] = (__bf16)v2;
}

// ------------------ norm helpers (degree over columns) ---------------------
// d[j] = sum_i H[i,j] - H[j,j] + addc + 1e-6
__global__ void colsum_kernel(const float* __restrict__ H, float* __restrict__ d,
                              float addc)
{
    int j = blockIdx.x * blockDim.x + threadIdx.x;
    if (j >= GN) return;
    float s = 0.f;
    for (int i = 0; i < GN; ++i) s += H[(size_t)i * GN + j];
    d[j] = s - H[(size_t)j * GN + j] + addc + 1e-6f;
}

// Hn[i,j] = (i==j ? 0 : H[i,j]) / d[j]
__global__ void make_hn_kernel(const float* __restrict__ H,
                               const float* __restrict__ d,
                               __bf16* __restrict__ Hn)
{
    size_t idx = (size_t)blockIdx.x * blockDim.x + threadIdx.x;
    if (idx >= (size_t)GN * GN) return;
    int i = (int)(idx / GN), j = (int)(idx % GN);
    float v = (i == j) ? 0.f : H[idx] / d[j];
    Hn[idx] = (__bf16)v;
}

// adj[p,q] = (p==q ? 1 : H2[q,p]) / d[p]   (LDS-tiled transpose read)
__global__ void make_adj_kernel(const float* __restrict__ H2,
                                const float* __restrict__ d,
                                __bf16* __restrict__ adj)
{
    __shared__ float tile[32][33];
    int bx = blockIdx.x * 32;   // p block
    int by = blockIdx.y * 32;   // q block
    int x = bx + threadIdx.x, y = by + threadIdx.y;
    tile[threadIdx.y][threadIdx.x] = H2[(size_t)y * GN + x];   // H2[q', p']
    __syncthreads();
    int p = bx + threadIdx.y, q = by + threadIdx.x;
    float num = (p == q) ? 1.f : tile[threadIdx.x][threadIdx.y];
    adj[(size_t)p * GN + q] = (__bf16)(num / d[p]);
}

// --------------------------- conversions -----------------------------------
__global__ void cvt_bf16_kernel(const float* __restrict__ s, __bf16* __restrict__ dst,
                                size_t n)
{
    size_t i = (size_t)blockIdx.x * blockDim.x + threadIdx.x;
    if (i < n) dst[i] = (__bf16)s[i];
}

// dst[C,R] = src[R,C]^T  (for weight matrices -> B^T operands)
__global__ void cvt_bf16_T_kernel(const float* __restrict__ s, __bf16* __restrict__ dst,
                                  int R, int Cc)
{
    size_t idx = (size_t)blockIdx.x * blockDim.x + threadIdx.x;
    if (idx >= (size_t)R * Cc) return;
    int r = (int)(idx / Cc), c = (int)(idx % Cc);
    dst[(size_t)c * R + r] = (__bf16)s[idx];
}

// ------------------------------ head ---------------------------------------
__global__ void zero1_kernel(float* p) { p[0] = 0.f; }

__global__ void head_kernel(const float* __restrict__ x4,   // [GN,128]
                            const float* __restrict__ w2,   // [128,10]
                            const float* __restrict__ b2,   // [10]
                            const int* __restrict__ tx, const int* __restrict__ tg,
                            float* __restrict__ out, int BTn)
{
    int t = blockIdx.x * blockDim.x + threadIdx.x;
    if (t >= BTn) return;
    int r = tx[t];
    float y[10];
    #pragma unroll
    for (int c = 0; c < 10; ++c) y[c] = b2[c];
    const float* xr = x4 + (size_t)r * 128;
    for (int k = 0; k < 128; ++k) {
        float xv = xr[k];
        const float* wr = w2 + k * 10;
        #pragma unroll
        for (int c = 0; c < 10; ++c) y[c] += xv * wr[c];
    }
    float* yo = out + 1 + (size_t)t * 10;
    float mx = y[0];
    #pragma unroll
    for (int c = 1; c < 10; ++c) mx = fmaxf(mx, y[c]);
    float se = 0.f;
    #pragma unroll
    for (int c = 0; c < 10; ++c) se += __expf(y[c] - mx);
    float lse = __logf(se);
    int ti = tg[t];
    float yt = y[0];
    #pragma unroll
    for (int c = 0; c < 10; ++c) {
        yo[c] = y[c];
        if (c == ti) yt = y[c];
    }
    float logp = yt - mx - lse;
    atomicAdd(out, -logp / (float)BTn);
}

// ------------------------------ driver -------------------------------------
extern "C" void kernel_launch(void* const* d_in, const int* in_sizes, int n_in,
                              void* d_out, int out_size, void* d_ws, size_t ws_size,
                              hipStream_t stream)
{
    (void)in_sizes; (void)n_in; (void)out_size; (void)ws_size;

    const float* A    = (const float*)d_in[0];
    const float* X    = (const float*)d_in[1];
    const float* g0w1 = (const float*)d_in[2];
    const float* g0w2 = (const float*)d_in[3];
    const float* g1w1 = (const float*)d_in[4];
    const float* gc1w = (const float*)d_in[5];  const float* gc1b = (const float*)d_in[6];
    const float* gc2w = (const float*)d_in[7];  const float* gc2b = (const float*)d_in[8];
    const float* gc3w = (const float*)d_in[9];  const float* gc3b = (const float*)d_in[10];
    const float* l1w  = (const float*)d_in[11]; const float* l1b  = (const float*)d_in[12];
    const float* l2w  = (const float*)d_in[13]; const float* l2b  = (const float*)d_in[14];
    const int*   tx   = (const int*)d_in[15];
    const int*   tg   = (const int*)d_in[16];
    float* out = (float*)d_out;

    const size_t NN = (size_t)GN * GN;

    // ---- workspace carve (aligned 256B) ----
    char* ws = (char*)d_ws;
    size_t cur = 0;
    auto alloc = [&](size_t bytes) -> char* {
        cur = (cur + 255) & ~(size_t)255;
        char* p = ws + cur;
        cur += bytes;
        return p;
    };
    __bf16* buf0 = (__bf16*)alloc(NN * 2);   // a  -> Hn -> GCN t/x scratch
    __bf16* buf1 = (__bf16*)alloc(NN * 2);   // bT -> adj
    __bf16* buf2 = (__bf16*)alloc(NN * 2);   // a1T -> Xb + weight transposes
    float*  F    = (float* )alloc(NN * 4);   // H  -> H2
    float*  csum = (float* )alloc(GN * 4);

    // GCN scratch inside buf0 (free after GEMM2)
    __bf16* t1T = buf0;                                 // [512 ,4096]
    __bf16* x1b = t1T + (size_t)512 * GN;               // [4096,512 ]
    __bf16* t2T = x1b + (size_t)GN * 512;               // [256 ,4096]
    __bf16* x2b = t2T + (size_t)256 * GN;               // [4096,256 ]
    __bf16* t3T = x2b + (size_t)GN * 256;               // [128 ,4096]
    __bf16* x3b = t3T + (size_t)128 * GN;               // [4096,128 ]
    float*  x4f = (float*)(x3b + (size_t)GN * 128);     // [4096,128 ] f32

    // scratch inside buf2 (free after GEMM2)
    __bf16* Xb  = buf2;                                 // [4096,1024]
    __bf16* W1T = Xb  + (size_t)GN * 1024;              // [512 ,1024]
    __bf16* W2T = W1T + (size_t)512 * 1024;             // [256 ,512 ]
    __bf16* W3T = W2T + (size_t)256 * 512;              // [128 ,256 ]
    __bf16* L1T = W3T + (size_t)128 * 256;              // [128 ,128 ]

    // specialized GEMM launchers (only the 4 combos the pipeline needs)
    auto gemm_f32 = [&](const __bf16* Am, const __bf16* Bt, float* Cf,
                        int M_, int N_, int K_) {
        dim3 g(N_ / BN, M_ / BM);
        gemm_bf16_wmma<0, true, false, false><<<g, 256, 0, stream>>>(
            Am, Bt, Cf, nullptr, nullptr, nullptr, M_, N_, K_);
    };
    auto gemm_bT = [&](const __bf16* Am, const __bf16* Bt, __bf16* CbT,
                       int M_, int N_, int K_) {
        dim3 g(N_ / BN, M_ / BM);
        gemm_bf16_wmma<0, false, false, true><<<g, 256, 0, stream>>>(
            Am, Bt, nullptr, nullptr, CbT, nullptr, M_, N_, K_);
    };
    auto gemm_lrelu_b = [&](const __bf16* Am, const __bf16* Bt, __bf16* Cb,
                            const float* bias, int M_, int N_, int K_) {
        dim3 g(N_ / BN, M_ / BM);
        gemm_bf16_wmma<1, false, true, false><<<g, 256, 0, stream>>>(
            Am, Bt, nullptr, Cb, nullptr, bias, M_, N_, K_);
    };
    auto gemm_relu_f = [&](const __bf16* Am, const __bf16* Bt, float* Cf,
                           const float* bias, int M_, int N_, int K_) {
        dim3 g(N_ / BN, M_ / BM);
        gemm_bf16_wmma<2, true, false, false><<<g, 256, 0, stream>>>(
            Am, Bt, Cf, nullptr, nullptr, bias, M_, N_, K_);
    };

    // 1) gtconv x3: a, b^T, a1^T (bf16)
    gtconv3_kernel<<<(int)(NN / 256), 256, 0, stream>>>(A, g0w1, g0w2, g1w1,
                                                        buf0, buf1, buf2);
    // 2) H = a @ b
    gemm_f32(buf0, buf1, F, GN, GN, GN);
    // 3) Hn = norm(H, add=False)
    colsum_kernel<<<GN / 256, 256, 0, stream>>>(F, csum, 0.f);
    make_hn_kernel<<<(int)(NN / 256), 256, 0, stream>>>(F, csum, buf0);
    // 4) H2 = Hn @ a1
    gemm_f32(buf0, buf2, F, GN, GN, GN);
    // 5) adj = norm(H2, add=True).T  (bf16, into buf1)
    colsum_kernel<<<GN / 256, 256, 0, stream>>>(F, csum, 1.f);
    {
        dim3 g(GN / 32, GN / 32), b(32, 32);
        make_adj_kernel<<<g, b, 0, stream>>>(F, csum, buf1);
    }
    // 6) operand prep for GCN
    cvt_bf16_kernel<<<(int)((GN * 1024 + 255) / 256), 256, 0, stream>>>(X, Xb, (size_t)GN * 1024);
    cvt_bf16_T_kernel<<<(1024 * 512 + 255) / 256, 256, 0, stream>>>(gc1w, W1T, 1024, 512);
    cvt_bf16_T_kernel<<<(512 * 256 + 255) / 256, 256, 0, stream>>>(gc2w, W2T, 512, 256);
    cvt_bf16_T_kernel<<<(256 * 128 + 255) / 256, 256, 0, stream>>>(gc3w, W3T, 256, 128);
    cvt_bf16_T_kernel<<<(128 * 128 + 255) / 256, 256, 0, stream>>>(l1w, L1T, 128, 128);
    // 7) GCN layer 1: x1 = lrelu(adj @ (X W1) + b1)
    gemm_bT(Xb, W1T, t1T, GN, 512, 1024);
    gemm_lrelu_b(buf1, t1T, x1b, gc1b, GN, 512, GN);
    // 8) layer 2
    gemm_bT(x1b, W2T, t2T, GN, 256, 512);
    gemm_lrelu_b(buf1, t2T, x2b, gc2b, GN, 256, GN);
    // 9) layer 3
    gemm_bT(x2b, W3T, t3T, GN, 128, 256);
    gemm_lrelu_b(buf1, t3T, x3b, gc3b, GN, 128, GN);
    // 10) x4 = relu(x3 @ lin1 + b)  (f32 out)
    gemm_relu_f(x3b, L1T, x4f, l1b, GN, 128, 128);
    // 11) head: gather, lin2, log_softmax, loss
    zero1_kernel<<<1, 1, 0, stream>>>(out);
    head_kernel<<<4, 256, 0, stream>>>(x4f, l2w, l2b, tx, tg, out, 1024);
}
